// SwinTransformerBlock_61890478736100
// MI455X (gfx1250) — matrix operand order repaired and checked by reading, not compiled, not run on hardware
//
#include <hip/hip_runtime.h>
#include <hip/hip_bf16.h>
#include <math.h>

typedef _Float16 h16;
typedef __attribute__((ext_vector_type(16))) _Float16 v16h;
typedef __attribute__((ext_vector_type(8)))  _Float16 v8h;
typedef __attribute__((ext_vector_type(8)))  float    v8f;

#define WMMA16(a, b, c) __builtin_amdgcn_wmma_f32_16x16x32_f16( \
    false, (a), false, (b), (short)0, (c), false, false)

// ---------------------------------------------------------------------------
// Fragment loaders (wave32; lane half h holds k = 16h..16h+15, same order for A and B), contiguous per lane
// so they lower to b128 vector loads.
__device__ inline v16h frag_a(const h16* base, int ld) {
  const int lane = threadIdx.x & 31;
  const h16* p = base + (size_t)(lane & 15) * ld + ((lane >> 4) << 4);
  v8h lo = *(const v8h*)(p);
  v8h hi = *(const v8h*)(p + 8);
  return __builtin_shufflevector(lo, hi, 0, 1, 2, 3, 4, 5, 6, 7,
                                 8, 9, 10, 11, 12, 13, 14, 15);
}

// B 32x16, element = base[col*ld + k]; lane reads 32B contiguous.
__device__ inline v16h frag_bT(const h16* base, int ld) {
  const int lane = threadIdx.x & 31;
  const h16* p = base + (size_t)(lane & 15) * ld + ((lane >> 4) << 4);
  return *(const v16h*)p;
}

__device__ inline v8f zero8() {
  v8f z;
#pragma unroll
  for (int i = 0; i < 8; ++i) z[i] = 0.f;
  return z;
}

// ---------------------------------------------------------------------------
// Weight pre-tiling: f32 row-major KxN -> f16 fragment tiles
__global__ __launch_bounds__(256)
void w_tile_kernel(const float* __restrict__ src, h16* __restrict__ dst,
                   int K, int N) {
  int i = blockIdx.x * 256 + threadIdx.x;
  if (i >= K * N) return;
  int k = i / N, n = i - k * N;
  size_t t = (size_t)(k >> 5) * (N >> 4) + (n >> 4);
  dst[t * 512 + (size_t)(n & 15) * 32 + (k & 31)] = (h16)src[i];
}

// ---------------------------------------------------------------------------
// Tiled WMMA GEMM with double-buffered LDS A staging.
// Block 128x128, BK=32, 8 waves (4x2), wave tile 32x64.
// mode 0: f16 out = acc+bias
// mode 1: QKV: q (col<512) scaled -> qk[row*1024+col]; k -> qk; v -> vT transposed
// mode 2: exact GELU; f16 out
// mode 3: f32 out = resid + acc + bias (final residual)
__global__ __launch_bounds__(256)
void gemm_wmma_kernel(const h16* __restrict__ A, const h16* __restrict__ Bt,
                      const float* __restrict__ bias, const float* resid,
                      void* out, h16* __restrict__ vT,
                      int M, int N, int K, int mode) {
  __shared__ __align__(16) h16 sA[2][128 * 40];   // 40 halves/row: conflict-free

  const int tid   = threadIdx.x;
  const int lane  = tid & 31;
  const int wave  = tid >> 5;
  const int waveM = wave >> 1;   // 0..3
  const int waveN = wave & 1;    // 0..1
  const int bm = blockIdx.y, bn = blockIdx.x;
  const int nTiles = N >> 4;

  const int aRow = tid >> 1, aKo = (tid & 1) * 16;
  const h16* aSrc = A + ((size_t)(bm * 128 + aRow)) * K + aKo;

  v8f acc[2][4];
#pragma unroll
  for (int i = 0; i < 2; ++i)
#pragma unroll
    for (int j = 0; j < 4; ++j) acc[i][j] = zero8();

  // prologue: stage K-step 0 into buffer 0
  {
    float4 d0 = *(const float4*)(aSrc);
    float4 d1 = *(const float4*)(aSrc + 8);
    float* dst = (float*)&sA[0][aRow * 40 + aKo];
    dst[0] = d0.x; dst[1] = d0.y; dst[2] = d0.z; dst[3] = d0.w;
    dst[4] = d1.x; dst[5] = d1.y; dst[6] = d1.z; dst[7] = d1.w;
  }

  const int kSteps = K >> 5;
  int buf = 0;
  for (int kc = 0; kc < kSteps; ++kc) {
    __syncthreads();

    // issue next A tile's global loads early (latency hidden under WMMAs)
    float4 n0, n1;
    const bool hasNext = (kc + 1 < kSteps);
    if (hasNext) {
      const h16* s = aSrc + (size_t)(kc + 1) * 32;
      n0 = *(const float4*)(s);
      n1 = *(const float4*)(s + 8);
      __builtin_prefetch(Bt + ((size_t)(kc + 1) * nTiles + bn * 8 + waveN * 4) * 512, 0, 1);
    }

    // hoist all four B fragments before the MMA group
    const h16* btBase = Bt + ((size_t)kc * nTiles + bn * 8 + waveN * 4) * 512;
    v16h bf[4];
#pragma unroll
    for (int jn = 0; jn < 4; ++jn) bf[jn] = frag_bT(btBase + jn * 512, 32);

    v16h a0 = frag_a(&sA[buf][(waveM * 32) * 40], 40);
    v16h a1 = frag_a(&sA[buf][(waveM * 32 + 16) * 40], 40);
#pragma unroll
    for (int jn = 0; jn < 4; ++jn) {
      acc[0][jn] = WMMA16(a0, bf[jn], acc[0][jn]);
      acc[1][jn] = WMMA16(a1, bf[jn], acc[1][jn]);
    }

    // commit next tile to the alternate buffer (prev reads done before sync)
    if (hasNext) {
      float* dst = (float*)&sA[buf ^ 1][aRow * 40 + aKo];
      dst[0] = n0.x; dst[1] = n0.y; dst[2] = n0.z; dst[3] = n0.w;
      dst[4] = n1.x; dst[5] = n1.y; dst[6] = n1.z; dst[7] = n1.w;
      buf ^= 1;
    }
  }

  const int colL = lane & 15;
  const int rowL = (lane >> 4) * 8;
#pragma unroll
  for (int mt = 0; mt < 2; ++mt) {
#pragma unroll
    for (int jn = 0; jn < 4; ++jn) {
      int col = bn * 128 + waveN * 64 + jn * 16 + colL;
      float bv = bias ? bias[col] : 0.f;
#pragma unroll
      for (int r = 0; r < 8; ++r) {
        int row = bm * 128 + waveM * 32 + mt * 16 + rowL + r;
        float v = acc[mt][jn][r] + bv;
        if (mode == 1) {
          if (col < 1024) {  // q (scaled) and k -> stride-1024 qk buffer
            if (col < 512) v *= 0.17677669529663687f;  // HD^-0.5
            ((h16*)out)[(size_t)row * 1024 + col] = (h16)v;
          } else {           // v -> transposed vT[win][head][d][n]
            int win = row >> 6, n = row & 63;
            int hd = col - 1024;
            vT[(((size_t)win * 16 + (hd >> 5)) * 32 + (hd & 31)) * 64 + n] = (h16)v;
          }
        } else if (mode == 2) {
          v = 0.5f * v * (1.f + erff(v * 0.70710678118654752f));
          ((h16*)out)[(size_t)row * N + col] = (h16)v;
        } else if (mode == 3) {
          size_t idx = (size_t)row * N + col;
          ((float*)out)[idx] = resid[idx] + v;
        } else {
          ((h16*)out)[(size_t)row * N + col] = (h16)v;
        }
      }
    }
  }
}

// ---------------------------------------------------------------------------
// Fused windowed attention: one block = (window, head); 4 waves x 16 rows.
// qk: (2048, 64, 1024) f16 (q pre-scaled). vT: (2048, 16, 32, 64) f16.
__global__ __launch_bounds__(128)
void attn_kernel(const h16* __restrict__ qk, const h16* __restrict__ vT,
                 const float* __restrict__ bias_table,
                 const int* __restrict__ relidx, const float* __restrict__ mask,
                 h16* __restrict__ o) {
  __shared__ __align__(16) h16 sP[4][16][72];

  const int win  = blockIdx.x >> 4;
  const int head = blockIdx.x & 15;
  const int wave = threadIdx.x >> 5;
  const int lane = threadIdx.x & 31;
  const int colL = lane & 15;
  const int rowL = (lane >> 4) * 8;

  const h16* qp = qk + (size_t)win * 64 * 1024 + head * 32;
  const h16* kp = qp + 512;
  const h16* vt = vT + (size_t)(win * 16 + head) * 32 * 64;

  // hoist q and all k fragments, then run the 4 score WMMAs back-to-back
  v16h qa = frag_a(qp + (size_t)(wave * 16) * 1024, 1024);
  v16h kb[4];
#pragma unroll
  for (int j = 0; j < 4; ++j)
    kb[j] = frag_bT(kp + (size_t)(j * 16) * 1024, 1024);

  v8f S[4];
#pragma unroll
  for (int j = 0; j < 4; ++j) S[j] = WMMA16(qa, kb[j], zero8());

  // relative-position bias + shifted-window mask; kill padded key columns
  const float* maskw = mask + (size_t)(win & 63) * 2401;
#pragma unroll
  for (int j = 0; j < 4; ++j) {
    int m = j * 16 + colL;
#pragma unroll
    for (int r = 0; r < 8; ++r) {
      int n = wave * 16 + rowL + r;
      if (m >= 49) {
        S[j][r] = -1e30f;
      } else {
        int nn = n < 49 ? n : 48;
        S[j][r] += bias_table[relidx[nn * 49 + m] * 16 + head] + maskw[nn * 49 + m];
      }
    }
  }

  // row softmax: 64 cols live in 4 accumulators x 16 lanes of each half-group
#pragma unroll
  for (int r = 0; r < 8; ++r) {
    float m0 = fmaxf(fmaxf(S[0][r], S[1][r]), fmaxf(S[2][r], S[3][r]));
    m0 = fmaxf(m0, __shfl_xor(m0, 1, 32));
    m0 = fmaxf(m0, __shfl_xor(m0, 2, 32));
    m0 = fmaxf(m0, __shfl_xor(m0, 4, 32));
    m0 = fmaxf(m0, __shfl_xor(m0, 8, 32));
    float s = 0.f;
#pragma unroll
    for (int j = 0; j < 4; ++j) { S[j][r] = __expf(S[j][r] - m0); s += S[j][r]; }
    s += __shfl_xor(s, 1, 32);
    s += __shfl_xor(s, 2, 32);
    s += __shfl_xor(s, 4, 32);
    s += __shfl_xor(s, 8, 32);
    float inv = 1.f / s;
#pragma unroll
    for (int j = 0; j < 4; ++j) S[j][r] *= inv;
  }

  // stage P through LDS: C-layout -> A-layout
#pragma unroll
  for (int j = 0; j < 4; ++j)
#pragma unroll
    for (int r = 0; r < 8; ++r)
      sP[wave][rowL + r][j * 16 + colL] = (h16)S[j][r];
  __syncthreads();

  v16h p0 = frag_a(&sP[wave][0][0], 72);    // K = 0..31
  v16h p1 = frag_a(&sP[wave][0][32], 72);   // K = 32..63

  // hoist all V fragments, then the 4 PV WMMAs
  v16h vf[2][2];
#pragma unroll
  for (int t = 0; t < 2; ++t) {
    vf[t][0] = frag_bT(vt + (size_t)(t * 16) * 64, 64);       // K rows 0..31
    vf[t][1] = frag_bT(vt + (size_t)(t * 16) * 64 + 32, 64);  // K rows 32..63
  }

  h16* ob = o + (size_t)win * 64 * 512 + head * 32;
#pragma unroll
  for (int t = 0; t < 2; ++t) {
    v8f a = WMMA16(p0, vf[t][0], zero8());
    a = WMMA16(p1, vf[t][1], a);
#pragma unroll
    for (int r = 0; r < 8; ++r) {
      int n = wave * 16 + rowL + r;
      ob[(size_t)n * 512 + t * 16 + colL] = (h16)a[r];
    }
  }
}

// ---------------------------------------------------------------------------
__device__ inline float block_sum256(float v) {
  __shared__ float sbuf[8];
#pragma unroll
  for (int m = 16; m >= 1; m >>= 1) v += __shfl_xor(v, m, 32);
  __syncthreads();
  if ((threadIdx.x & 31) == 0) sbuf[threadIdx.x >> 5] = v;
  __syncthreads();
  float s = 0.f;
#pragma unroll
  for (int i = 0; i < 8; ++i) s += sbuf[i];
  return s;
}

// LN1 + cyclic shift(-3,-3) + window partition into padded (2048,64,512) f16
__global__ __launch_bounds__(256)
void ln1_window_kernel(const float* __restrict__ x, const float* __restrict__ g,
                       const float* __restrict__ bta, h16* __restrict__ wins) {
  const int idx = blockIdx.x;        // win*64 + n
  const int win = idx >> 6, n = idx & 63;
  h16* op = wins + (size_t)idx * 512;
  if (n >= 49) {                     // zero padding rows (uniform per block)
    for (int c = threadIdx.x; c < 512; c += 256) op[c] = (h16)0.f;
    return;
  }
  const int b = win >> 6, w = win & 63;
  const int r = n / 7, cc = n - r * 7;
  const int sh = ((w >> 3) * 7 + r + 3) % 56;    // roll(-3) => src = dst+3
  const int sw = ((w & 7) * 7 + cc + 3) % 56;
  const float* row = x + ((size_t)b * 3136 + sh * 56 + sw) * 512;

  float a0 = row[threadIdx.x], a1 = row[threadIdx.x + 256];
  float mu = block_sum256(a0 + a1) * (1.f / 512.f);
  float d0 = a0 - mu, d1 = a1 - mu;
  float var = block_sum256(d0 * d0 + d1 * d1) * (1.f / 512.f);
  float rs = rsqrtf(var + 1e-5f);
  op[threadIdx.x]       = (h16)(d0 * rs * g[threadIdx.x] + bta[threadIdx.x]);
  op[threadIdx.x + 256] = (h16)(d1 * rs * g[threadIdx.x + 256] + bta[threadIdx.x + 256]);
}

// window reverse + roll(+3,+3) + residual add (writes x1 = shortcut + attnproj)
__global__ __launch_bounds__(256)
void reverse_add_kernel(const float* __restrict__ x, const h16* __restrict__ proj,
                        float* __restrict__ out) {
  size_t i = (size_t)blockIdx.x * 256 + threadIdx.x;   // 51,380,224 elems
  int c = (int)(i & 511);
  size_t pix = i >> 9;
  int b = (int)(pix / 3136);
  int hw = (int)(pix - (size_t)b * 3136);
  int hh = hw / 56, ww = hw - hh * 56;
  int hs = hh - 3; if (hs < 0) hs += 56;
  int ws = ww - 3; if (ws < 0) ws += 56;
  int win = b * 64 + (hs / 7) * 8 + (ws / 7);
  int n = (hs % 7) * 7 + (ws % 7);
  out[i] = x[i] + (float)proj[((size_t)win * 64 + n) * 512 + c];
}

__global__ __launch_bounds__(256)
void ln2_kernel(const float* __restrict__ xin, const float* __restrict__ g,
                const float* __restrict__ bta, h16* __restrict__ out) {
  const float* row = xin + (size_t)blockIdx.x * 512;
  h16* op = out + (size_t)blockIdx.x * 512;
  float a0 = row[threadIdx.x], a1 = row[threadIdx.x + 256];
  float mu = block_sum256(a0 + a1) * (1.f / 512.f);
  float d0 = a0 - mu, d1 = a1 - mu;
  float var = block_sum256(d0 * d0 + d1 * d1) * (1.f / 512.f);
  float rs = rsqrtf(var + 1e-5f);
  op[threadIdx.x]       = (h16)(d0 * rs * g[threadIdx.x] + bta[threadIdx.x]);
  op[threadIdx.x + 256] = (h16)(d1 * rs * g[threadIdx.x + 256] + bta[threadIdx.x + 256]);
}

// ---------------------------------------------------------------------------
extern "C" void kernel_launch(void* const* d_in, const int* in_sizes, int n_in,
                              void* d_out, int out_size, void* d_ws, size_t ws_size,
                              hipStream_t stream) {
  const float* x     = (const float*)d_in[0];
  const float* n1g   = (const float*)d_in[1];
  const float* n1b   = (const float*)d_in[2];
  const float* qkvw  = (const float*)d_in[3];
  const float* qkvb  = (const float*)d_in[4];
  const float* relbt = (const float*)d_in[5];
  const float* projw = (const float*)d_in[6];
  const float* projb = (const float*)d_in[7];
  const float* n2g   = (const float*)d_in[8];
  const float* n2b   = (const float*)d_in[9];
  const float* fc1w  = (const float*)d_in[10];
  const float* fc1b  = (const float*)d_in[11];
  const float* fc2w  = (const float*)d_in[12];
  const float* fc2b  = (const float*)d_in[13];
  const int*   reli  = (const int*)d_in[14];
  const float* amask = (const float*)d_in[15];
  float* xout = (float*)d_out;

  // workspace (halves): tiled weights (6.3MB) | bufA 128MB | bufB 392MB
  h16* wQKV = (h16*)d_ws;
  h16* wProj = wQKV + 512 * 1536;
  h16* wFc1 = wProj + 512 * 512;
  h16* wFc2 = wFc1 + 512 * 2048;
  h16* bufA = (h16*)d_ws + (size_t)(4u << 20);     // +8MB
  h16* bufB = bufA + (size_t)67108864;             // +128MB
  h16* qkB  = bufB;                                // 2048*64*1024 halves
  h16* vTB  = bufB + (size_t)134217728;            // 2048*16*32*64 halves

  // pre-tile weights fp32 -> f16 fragment layout
  w_tile_kernel<<<(512 * 1536 + 255) / 256, 256, 0, stream>>>(qkvw, wQKV, 512, 1536);
  w_tile_kernel<<<(512 * 512 + 255) / 256, 256, 0, stream>>>(projw, wProj, 512, 512);
  w_tile_kernel<<<(512 * 2048 + 255) / 256, 256, 0, stream>>>(fc1w, wFc1, 512, 2048);
  w_tile_kernel<<<(2048 * 512 + 255) / 256, 256, 0, stream>>>(fc2w, wFc2, 2048, 512);

  // LN1 + shift + partition -> bufA (2048 x 64 x 512 f16, padded)
  ln1_window_kernel<<<131072, 256, 0, stream>>>(x, n1g, n1b, bufA);

  // QKV: (131072 x 512) @ (512 x 1536); q,k -> qkB (stride 1024), v -> vTB
  gemm_wmma_kernel<<<dim3(12, 1024), 256, 0, stream>>>(
      bufA, wQKV, qkvb, nullptr, qkB, vTB, 131072, 1536, 512, 1);

  // fused attention -> bufA (o, 2048 x 64 x 512 f16)
  attn_kernel<<<32768, 128, 0, stream>>>(qkB, vTB, relbt, reli, amask, bufA);

  // proj: (131072 x 512) @ (512 x 512) -> bufB (f16, +bias)
  gemm_wmma_kernel<<<dim3(4, 1024), 256, 0, stream>>>(
      bufA, wProj, projb, nullptr, bufB, nullptr, 131072, 512, 512, 0);

  // reverse + roll back + residual -> d_out (x1, f32)
  reverse_add_kernel<<<200704, 256, 0, stream>>>(x, bufB, xout);

  // LN2 -> bufA (100352 x 512 f16)
  ln2_kernel<<<100352, 256, 0, stream>>>(xout, n2g, n2b, bufA);

  // fc1 + exact GELU: (100352 x 512) @ (512 x 2048) -> bufB
  gemm_wmma_kernel<<<dim3(16, 784), 256, 0, stream>>>(
      bufA, wFc1, fc1b, nullptr, bufB, nullptr, 100352, 2048, 512, 2);

  // fc2 + residual: (100352 x 2048) @ (2048 x 512) + x1 -> d_out (f32)
  gemm_wmma_kernel<<<dim3(4, 784), 256, 0, stream>>>(
      bufB, wFc2, fc2b, xout, d_out, nullptr, 100352, 512, 2048, 3);
}
